// FeatureAttnNet_62053687493032
// MI455X (gfx1250) — matrix operand order repaired and checked by reading, not compiled, not run on hardware
//
#include <hip/hip_runtime.h>
#include <hip/hip_bf16.h>
#include <stdint.h>

// Sizes from the reference
#define Bsz   128
#define NENC  2048
#define NCMD  512
#define Dd    64
#define NF    1024
#define Hh    8
#define Mout  32768   // D*NF/2 per GEMM

typedef __attribute__((ext_vector_type(8)))  float    v8f;
typedef __attribute__((ext_vector_type(16))) __bf16   v16bf;
typedef __attribute__((ext_vector_type(8)))  __bf16   v8bf;
typedef __attribute__((ext_vector_type(2)))  __bf16   v2bf;
typedef __attribute__((ext_vector_type(4)))  uint32_t u32x4;
typedef __attribute__((ext_vector_type(8)))  uint32_t u32x8;

__device__ __forceinline__ v16bf cat8(v8bf lo, v8bf hi) {
    v16bf r;
#pragma unroll
    for (int i = 0; i < 8; ++i) { r[i] = lo[i]; r[i + 8] = hi[i]; }
    return r;
}

// ---------------------------------------------------------------------------
// Kernel A: y = LayerNorm(GELU([feature;hidden] @ W^T)) -> [B,NF,D] bf16.
// grid = (512, 2). One workgroup = all 128 batch rows x one 64-col LN group,
// so the 536 MB of weights stream from HBM exactly once.  Double-buffered
// LDS + register prefetch hides WMMA under the weight stream.
// ---------------------------------------------------------------------------
__global__ __launch_bounds__(256) void ka_gemm_gelu_ln(
    const float* __restrict__ feat, const float* __restrict__ hid,
    const float* __restrict__ wz,   const float* __restrict__ wh,
    const float* __restrict__ ln_g, const float* __restrict__ ln_b,
    __bf16* __restrict__ y)
{
    const int bx    = blockIdx.x;           // 0..511
    const int which = blockIdx.y;           // 0 = z (feature), 1 = h (hidden)
    const float* act = which ? hid : feat;
    const float* w   = which ? wh  : wz;
    const int m0 = bx * 64;
    const int f  = bx + which * 512;

    __shared__ __align__(16) __bf16 lA[2][128 * 72];   // act chunk, pad 72
    __shared__ __align__(16) __bf16 lB[2][64 * 72];    // weight chunk [n][k]
    __shared__ float stage[8 * 16 * 65];
    __shared__ float muA[128], rsA[128];

    const int tid = threadIdx.x;
    const int wv  = tid >> 5;
    const int ln  = tid & 31;
    const int hl  = ln >> 4;
    const int l16 = ln & 15;

    // per-thread staging assignments
    const int ar = tid >> 1, aks = (tid & 1) * 32;     // A: 32 floats
    const int br = tid >> 2, bks = (tid & 3) * 16;     // B: 16 floats
    const float* aptr = act + (size_t)ar * NENC + aks;
    const float* bptr = w + (size_t)(m0 + br) * NENC + bks;

    float ra[32], rb[16];
#pragma unroll
    for (int j = 0; j < 32; ++j) ra[j] = aptr[j];      // prefetch chunk 0
#pragma unroll
    for (int j = 0; j < 16; ++j) rb[j] = bptr[j];

    v8f acc[4] = {};

    for (int c = 0; c < NENC / 64; ++c) {
        const int cur = c & 1;
        { // stage prefetched regs -> LDS (fp32 -> bf16)
            __bf16* da = lA[cur] + ar * 72 + aks;
#pragma unroll
            for (int j = 0; j < 32; j += 2) {
                v2bf p; p[0] = (__bf16)ra[j]; p[1] = (__bf16)ra[j + 1];
                *(v2bf*)(da + j) = p;
            }
            __bf16* db = lB[cur] + br * 72 + bks;
#pragma unroll
            for (int j = 0; j < 16; j += 2) {
                v2bf p; p[0] = (__bf16)rb[j]; p[1] = (__bf16)rb[j + 1];
                *(v2bf*)(db + j) = p;
            }
        }
        __syncthreads();
        if (c + 1 < NENC / 64) {  // issue next chunk's global loads now
            const float* ap = aptr + (size_t)(c + 1) * 64;
            const float* bp = bptr + (size_t)(c + 1) * 64;
#pragma unroll
            for (int j = 0; j < 32; ++j) ra[j] = ap[j];
#pragma unroll
            for (int j = 0; j < 16; ++j) rb[j] = bp[j];
        }
        // compute from buffer `cur` while prefetch is in flight
#pragma unroll
        for (int ks = 0; ks < 64; ks += 32) {
            int arow = wv * 16 + l16;
            int ka0  = ks + hl * 16;
            v16bf a = cat8(*(const v8bf*)(lA[cur] + arow * 72 + ka0),
                           *(const v8bf*)(lA[cur] + arow * 72 + ka0 + 8));
#pragma unroll
            for (int t = 0; t < 4; ++t) {
                int n   = t * 16 + l16;
                int kb0 = ks + hl * 16;
                v16bf bf = cat8(*(const v8bf*)(lB[cur] + n * 72 + kb0),
                                *(const v8bf*)(lB[cur] + n * 72 + kb0 + 8));
                acc[t] = __builtin_amdgcn_wmma_f32_16x16x32_bf16(
                    false, a, false, bf, (short)0, acc[t], false, false);
            }
        }
        __syncthreads();
    }

    // exact GELU, stage, LayerNorm over the 64-col group, write y bf16
#pragma unroll
    for (int t = 0; t < 4; ++t)
#pragma unroll
        for (int r = 0; r < 8; ++r) {
            float x = acc[t][r];
            float g = 0.5f * x * (1.0f + erff(x * 0.70710678118654752f));
            int row = wv * 16 + r + hl * 8;
            stage[row * 65 + t * 16 + l16] = g;
        }
    __syncthreads();
    if (tid < 128) {
        const float* rp = stage + tid * 65;
        float s = 0.f;
        for (int j = 0; j < 64; ++j) s += rp[j];
        float mu = s * (1.f / 64.f), v = 0.f;
        for (int j = 0; j < 64; ++j) { float d = rp[j] - mu; v += d * d; }
        muA[tid] = mu;
        rsA[tid] = rsqrtf(v * (1.f / 64.f) + 1e-5f);
    }
    __syncthreads();
    {
        int row = tid >> 1, c0 = (tid & 1) * 32;
        float mu = muA[row], rs = rsA[row];
        __bf16* dst = y + ((size_t)row * NF + f) * Dd + c0;
        const float* sp = stage + row * 65 + c0;
#pragma unroll
        for (int j = 0; j < 32; j += 2) {
            int d0 = c0 + j;
            v2bf p;
            p[0] = (__bf16)((sp[j]     - mu) * rs * ln_g[d0]     + ln_b[d0]);
            p[1] = (__bf16)((sp[j + 1] - mu) * rs * ln_g[d0 + 1] + ln_b[d0 + 1]);
            *(v2bf*)(dst + j) = p;
        }
    }
}

// ---------------------------------------------------------------------------
// Kernel B: qk_vec[b, h*64+j] = sum_d (cmd_b . q_w[h*64+d,:]) * k_w[h*64+d, j]
// ---------------------------------------------------------------------------
__global__ __launch_bounds__(256) void kb_qk(
    const float* __restrict__ cmd, const float* __restrict__ qw,
    const float* __restrict__ kw,  float* __restrict__ qk)
{
    int b = blockIdx.x, tid = threadIdx.x;
    __shared__ float c[NCMD], Q[512];
    c[tid]       = cmd[(size_t)b * NCMD + tid];
    c[tid + 256] = cmd[(size_t)b * NCMD + tid + 256];
    __syncthreads();
    for (int o = tid; o < 512; o += 256) {
        const float* wr = qw + (size_t)o * NCMD;
        float s = 0.f;
        for (int k = 0; k < NCMD; ++k) s += c[k] * wr[k];
        Q[o] = s;
    }
    __syncthreads();
    for (int j = tid; j < 512; j += 256) {
        int h = j >> 6, jj = j & 63;
        const float* qb = Q + h * 64;
        float s = 0.f;
#pragma unroll 8
        for (int d = 0; d < 64; ++d) s += qb[d] * kw[(size_t)(h * 64 + d) * 64 + jj];
        qk[(size_t)b * 512 + j] = s;
    }
}

// ---------------------------------------------------------------------------
// Kernel C: per batch b.  Tensor Data Mover stages the whole y[b] tile
// (1024x64 bf16) into LDS with hardware padding to a stride-72 layout
// (pad 4 DWORDs every 32 DWORDs -> 144B rows, bank-conflict free), then:
//   WMMA V columns for the |V| gate + fused QK -> softmax -> S = v_w_h (A^T y)
// ---------------------------------------------------------------------------
__global__ __launch_bounds__(256) void kc_attn(
    const __bf16* __restrict__ y, const float* __restrict__ qkv,
    const float* __restrict__ vw, float* __restrict__ out)
{
    int b = blockIdx.x, tid = threadIdx.x;
    int wv = tid >> 5, ln = tid & 31, hl = ln >> 4, l16 = ln & 15;

    __shared__ __align__(16) __bf16 lY[NF * 72];    // y[b] staged by TDM, 144 KB
    __shared__ __align__(16) __bf16 lV[512 * 72];   // v_w bf16, 72 KB
    __shared__ float lQ[512];
    __shared__ float lL[Hh * NF];
    __shared__ float lU[Hh * Dd];

    // ---- Tensor Data Mover: y[b] (global, row stride 64*2B) -> lY (padded) ----
    if (wv == 0) {
        uint64_t ga  = (uint64_t)(uintptr_t)(y + (size_t)b * NF * Dd);
        uint32_t lds = (uint32_t)(uintptr_t)(&lY[0]);   // low 32 bits = LDS offset
        u32x4 g0;
        g0[0] = 1u;                                   // count=1, user descriptor
        g0[1] = lds;                                  // lds_addr
        g0[2] = (uint32_t)ga;                         // global_addr[31:0]
        g0[3] = (uint32_t)(ga >> 32) | (2u << 30);    // global_addr[56:32] | type=2
        u32x8 g1;
        g1[0] = (1u << 16)        // data_size = 2 bytes
              | (1u << 20)        // pad_enable
              | (4u << 22)        // pad_interval: every 32 DWORDs (=128B row)
              | (3u << 25);       // pad_amount: 4 DWORDs (=16B) -> 144B stride
        g1[1] = 64u << 16;        // tensor_dim0 = 64  (bits 79:48, low half)
        g1[2] = 1024u << 16;      // tensor_dim1 = 1024 (bits 111:80, low half)
        g1[3] = 64u << 16;        // tile_dim0 = 64 (bits 127:112)
        g1[4] = 1024u;            // tile_dim1 = 1024 (bits 143:128)
        g1[5] = 64u;              // tensor_dim0_stride = 64 (bits 207:160 low)
        g1[6] = 0u;
        g1[7] = 0u;
        asm volatile("tensor_load_to_lds %0, %1"
                     :: "s"(g0), "s"(g1) : "memory");
    }

    { // stage v_w as bf16 (fp32 source, must convert in-register)
        int r0 = tid * 2;
        for (int rr = 0; rr < 2; ++rr) {
            const float* src = vw + (size_t)(r0 + rr) * 64;
            __bf16* dst = lV + (r0 + rr) * 72;
#pragma unroll
            for (int j = 0; j < 64; j += 2) {
                v2bf p; p[0] = (__bf16)src[j]; p[1] = (__bf16)src[j + 1];
                *(v2bf*)(dst + j) = p;
            }
        }
    }
    lQ[tid]       = qkv[(size_t)b * 512 + tid];
    lQ[tid + 256] = qkv[(size_t)b * 512 + tid + 256];

    if (wv == 0) __builtin_amdgcn_s_wait_tensorcnt(0);
    __syncthreads();

    // logits[h][f] = (|V| gate) * QK / sqrt(D); all operands now in LDS
    for (int fb = wv; fb < NF / 16; fb += 8) {
        int f0 = fb * 16;
        const __bf16* ybr = lY + (size_t)(f0 + l16) * 72;
        int kb = hl * 16;
        v16bf B0 = cat8(*(const v8bf*)(ybr + kb),      *(const v8bf*)(ybr + kb + 8));
        v16bf B1 = cat8(*(const v8bf*)(ybr + 32 + kb), *(const v8bf*)(ybr + 32 + kb + 8));
        for (int h = 0; h < Hh; ++h) {
            v8f acc[4] = {};
#pragma unroll
            for (int t = 0; t < 4; ++t) {
                int arow = h * 64 + t * 16 + l16;
                int ka0 = hl * 16;
                v16bf a0 = cat8(*(const v8bf*)(lV + arow * 72 + ka0),
                                *(const v8bf*)(lV + arow * 72 + ka0 + 8));
                acc[t] = __builtin_amdgcn_wmma_f32_16x16x32_bf16(
                    false, a0, false, B0, (short)0, acc[t], false, false);
                v16bf a1 = cat8(*(const v8bf*)(lV + arow * 72 + 32 + ka0),
                                *(const v8bf*)(lV + arow * 72 + 32 + ka0 + 8));
                acc[t] = __builtin_amdgcn_wmma_f32_16x16x32_bf16(
                    false, a1, false, B1, (short)0, acc[t], false, false);
            }
            float s = 0.f;
#pragma unroll
            for (int t = 0; t < 4; ++t)
#pragma unroll
                for (int r = 0; r < 8; ++r) s += fabsf(acc[t][r]);
            s += __shfl_xor(s, 16);
            float mv = s * (1.f / 64.f);

            float qk = 0.f;
            const float* qh = lQ + h * 64;
            int k0 = hl * 16;
#pragma unroll
            for (int i = 0; i < 16; ++i) qk += qh[k0 + i]      * (float)B0[i];
#pragma unroll
            for (int i = 0; i < 16; ++i) qk += qh[32 + k0 + i] * (float)B1[i];
            qk += __shfl_xor(qk, 16);

            lL[h * NF + f0 + l16] = mv * qk * 0.125f;
        }
    }
    __syncthreads();

    // softmax over NF + u = A^T y : wave wv owns head h = wv
    {
        int h = wv;
        float m = -3.4e38f;
        for (int i = ln; i < NF; i += 32) m = fmaxf(m, lL[h * NF + i]);
        for (int d = 16; d; d >>= 1) m = fmaxf(m, __shfl_xor(m, d));
        float s = 0.f;
        for (int i = ln; i < NF; i += 32) {
            float e = __expf(lL[h * NF + i] - m);
            lL[h * NF + i] = e; s += e;
        }
        for (int d = 16; d; d >>= 1) s += __shfl_xor(s, d);
        float inv = 1.f / s;

        float u0 = 0.f, u1 = 0.f;
        int d0 = ln * 2;
        for (int ff = 0; ff < NF; ++ff) {
            float a = lL[h * NF + ff] * inv;
            v2bf yp = *(const v2bf*)(lY + (size_t)ff * 72 + d0);
            u0 += a * (float)yp[0];
            u1 += a * (float)yp[1];
        }
        lU[h * 64 + d0] = u0; lU[h * 64 + d0 + 1] = u1;
    }
    __syncthreads();
    { // S[b,h,d] = v_w_h[d,:] . u[h,:]
        int h = wv;
#pragma unroll
        for (int half = 0; half < 2; ++half) {
            int d = ln + half * 32;
            const __bf16* vr = lV + (size_t)(h * 64 + d) * 72;
            const float*  ur = lU + h * 64;
            float s = 0.f;
#pragma unroll 8
            for (int k = 0; k < 64; ++k) s += (float)vr[k] * ur[k];
            out[(size_t)b * 512 + h * 64 + d] = s;
        }
    }
}

// ---------------------------------------------------------------------------
extern "C" void kernel_launch(void* const* d_in, const int* in_sizes, int n_in,
                              void* d_out, int out_size, void* d_ws, size_t ws_size,
                              hipStream_t stream) {
    const float* feat = (const float*)d_in[0];
    const float* hid  = (const float*)d_in[1];
    const float* cmd  = (const float*)d_in[2];
    const float* wz   = (const float*)d_in[3];
    const float* wh   = (const float*)d_in[4];
    const float* qw   = (const float*)d_in[5];
    const float* kw   = (const float*)d_in[6];
    const float* vw   = (const float*)d_in[7];
    const float* lng  = (const float*)d_in[8];
    const float* lnb  = (const float*)d_in[9];
    float* out = (float*)d_out;

    __bf16* yws  = (__bf16*)d_ws;                                     // 16.8 MB
    float*  qkws = (float*)((char*)d_ws + (size_t)Bsz * NF * Dd * 2); // 256 KB

    ka_gemm_gelu_ln<<<dim3(Mout / 64, 2), 256, 0, stream>>>(
        feat, hid, wz, wh, lng, lnb, yws);
    kb_qk<<<Bsz, 256, 0, stream>>>(cmd, qw, kw, qkws);
    kc_attn<<<Bsz, 256, 0, stream>>>(yws, qkws, vw, out);
}
